// BooleanReservoir_3255585210786
// MI455X (gfx1250) — 8-bit matrix operand order repaired and checked by reading, compiled, not run on hardware
//
#include <hip/hip_runtime.h>
#include <math.h>

#define N_NODES 20000
#define N_INPUT 64
#define K_MAX   12
#define BITS    64
#define N_OUT   10
#define BATCH   64
#define TSTEPS  128
#define N_FEAT  (N_NODES - N_INPUT)
#define SROW    64

typedef __attribute__((ext_vector_type(8))) int   v8i;
typedef __attribute__((ext_vector_type(8))) float v8f;
typedef __attribute__((ext_vector_type(2))) float v2f;

__global__ void k_input_wmma(const int* __restrict__ x, const int* __restrict__ w_in,
                             unsigned char* __restrict__ u_all) {
  const int tid  = blockIdx.x * blockDim.x + threadIdx.x;
  const int wave = tid >> 5;
  const int lane = threadIdx.x & 31;
  const int rowTile = wave >> 2;
  const int colTile = wave & 3;
  const int Rbase = rowTile * 16;
  const int Nbase = colTile * 16;
  const int l15 = lane & 15;
  const bool hi = lane >= 16;

  const int row = Rbase + l15;
  const int* __restrict__ xr = x + row * BITS;
  const int kbaseA[8] = {0, 4, 16, 20, 32, 36, 48, 52};
  v8i A;
#pragma unroll
  for (int v = 0; v < 8; ++v) {
    const int kb = kbaseA[v] + (hi ? 8 : 0);
    int w = 0;
#pragma unroll
    for (int e = 0; e < 4; ++e) w |= (xr[kb + e] & 1) << (8 * e);
    A[v] = w;
  }

  const int col = Nbase + l15;
  v8i Bm;
#pragma unroll
  for (int v = 0; v < 8; ++v) {
    const int kb = (v >> 1) * 16 + (hi ? 8 : 0) + (v & 1) * 4;
    int w = 0;
#pragma unroll
    for (int e = 0; e < 4; ++e) w |= (w_in[(kb + e) * N_INPUT + col] & 1) << (8 * e);
    Bm[v] = w;
  }

  v8i C = {};
  v8i D = __builtin_amdgcn_wmma_i32_16x16x64_iu8(false, A, false, Bm, C, false, false);

#pragma unroll
  for (int r = 0; r < 8; ++r) {
    const int R = Rbase + r + (hi ? 8 : 0);
    const int b = R >> 7;
    const int t = R & 127;
    u_all[t * (N_INPUT * BATCH) + col * BATCH + b] = (D[r] > 0) ? 1 : 0;
  }
}

__global__ void k_pack_lut(const int* __restrict__ lut, unsigned int* __restrict__ lp) {
  const int tid = blockIdx.x * blockDim.x + threadIdx.x;
  if (tid >= N_NODES * 128) return;
  const int* __restrict__ src = lut + (size_t)tid * 32;
  unsigned int v = 0;
#pragma unroll
  for (int i = 0; i < 32; ++i) v |= (unsigned)(src[i] & 1) << i;
  lp[tid] = v;
}

__global__ void k_prep_adj(const int* __restrict__ adj, const int* __restrict__ mask,
                           int* __restrict__ eff, unsigned char* __restrict__ flag) {
  const int n = blockIdx.x * blockDim.x + threadIdx.x;
  if (n >= N_NODES) return;
  int any = 0;
#pragma unroll
  for (int j = 0; j < K_MAX; ++j) {
    const int m = mask[n * K_MAX + j];
    any |= m;
    eff[n * K_MAX + j] = m ? adj[n * K_MAX + j] : N_NODES;
  }
  flag[n] = any ? 0 : 1;
}

__global__ void k_init(const int* __restrict__ init_states, const unsigned char* __restrict__ u_all,
                       unsigned char* __restrict__ sA, unsigned char* __restrict__ sB) {
  const int tid = blockIdx.x * blockDim.x + threadIdx.x;
  if (tid >= (N_NODES + 1) * SROW) return;
  const int n = tid >> 6, b = tid & 63;
  unsigned char v;
  if (n == N_NODES)      v = 0;
  else if (n < N_INPUT)  v = u_all[n * BATCH + b];
  else                   v = (unsigned char)(init_states[n] & 1);
  sA[tid] = v;
  sB[tid] = v;
}

__global__ void k_zero_stage(float* __restrict__ stage) {
  const int tid = blockIdx.x * blockDim.x + threadIdx.x;
  if (tid < BATCH * 16) stage[tid] = 0.0f;
}

__global__ void __launch_bounds__(256)
k_step(const unsigned char* __restrict__ prev, unsigned char* __restrict__ cur,
       const int* __restrict__ eff, const unsigned char* __restrict__ flag,
       const unsigned int* __restrict__ lp, const unsigned char* __restrict__ u_all,
       int t) {
  const int tid = blockIdx.x * blockDim.x + threadIdx.x;
  const int n = tid >> 6, b = tid & 63;

  if (n < N_INPUT) {
    const int tn = (t + 1 < TSTEPS) ? (t + 1) : (TSTEPS - 1);
    cur[tid] = u_all[tn * (N_INPUT * BATCH) + n * BATCH + b];
    return;
  }
  if (flag[n]) { cur[tid] = prev[tid]; return; }

  const int4* __restrict__ e4 = (const int4*)(eff + n * K_MAX);
  const int4 e0 = e4[0], e1 = e4[1], e2 = e4[2];

  int idx = 0;
  idx = (idx << 1) | prev[(e0.x << 6) + b];
  idx = (idx << 1) | prev[(e0.y << 6) + b];
  idx = (idx << 1) | prev[(e0.z << 6) + b];
  idx = (idx << 1) | prev[(e0.w << 6) + b];
  idx = (idx << 1) | prev[(e1.x << 6) + b];
  idx = (idx << 1) | prev[(e1.y << 6) + b];
  idx = (idx << 1) | prev[(e1.z << 6) + b];
  idx = (idx << 1) | prev[(e1.w << 6) + b];
  idx = (idx << 1) | prev[(e2.x << 6) + b];
  idx = (idx << 1) | prev[(e2.y << 6) + b];
  idx = (idx << 1) | prev[(e2.z << 6) + b];
  idx = (idx << 1) | prev[(e2.w << 6) + b];

  cur[tid] = (lp[n * 128 + (idx >> 5)] >> (idx & 31)) & 1;
}

__global__ void __launch_bounds__(256)
k_readout(const unsigned char* __restrict__ states, const float* __restrict__ W,
          float* __restrict__ stage) {
  const int tid  = blockIdx.x * blockDim.x + threadIdx.x;
  const int wave = tid >> 5;
  const int lane = threadIdx.x & 31;
  const int Mtile = wave & 3;
  const int chunk = wave >> 2;
  const int Mbase = Mtile * 16;
  const int l15 = lane & 15;
  const bool hi = lane >= 16;
  const int rowA = Mbase + l15;
  const int kOff = hi ? 2 : 0;
  const int o = l15;
  const int k0 = chunk * 89 * 4;

  v8f C = {};
  for (int it = 0; it < 89; ++it) {
    const int k = k0 + it * 4 + kOff;
    v2f A, Bv;
    A.x = (float)states[(N_INPUT + k)     * SROW + rowA];
    A.y = (float)states[(N_INPUT + k + 1) * SROW + rowA];
    Bv.x = (o < N_OUT) ? W[o * N_FEAT + k]     : 0.0f;
    Bv.y = (o < N_OUT) ? W[o * N_FEAT + k + 1] : 0.0f;
    C = __builtin_amdgcn_wmma_f32_16x16x4_f32(false, A, false, Bv, (short)0, C,
                                              false, false);
  }
#pragma unroll
  for (int r = 0; r < 8; ++r) {
    const int bidx = Mbase + r + (hi ? 8 : 0);
    atomicAdd(&stage[bidx * 16 + o], C[r]);
  }
}

__global__ void k_final(const float* __restrict__ stage, const float* __restrict__ bias,
                        float* __restrict__ out) {
  const int tid = blockIdx.x * blockDim.x + threadIdx.x;
  if (tid >= BATCH * N_OUT) return;
  const int b = tid / N_OUT, o = tid % N_OUT;
  const float v = stage[b * 16 + o] + bias[o];
  out[tid] = 1.0f / (1.0f + __expf(-v));
}

static inline size_t align256(size_t x) { return (x + 255) & ~(size_t)255; }

extern "C" void kernel_launch(void* const* d_in, const int* in_sizes, int n_in,
                              void* d_out, int out_size, void* d_ws, size_t ws_size,
                              hipStream_t stream) {
  const int*   x           = (const int*)  d_in[0];
  const int*   w_in        = (const int*)  d_in[1];
  const int*   adj_list    = (const int*)  d_in[2];
  const int*   adj_mask    = (const int*)  d_in[3];
  const int*   lut         = (const int*)  d_in[4];
  const int*   init_states = (const int*)  d_in[5];
  const float* W_out       = (const float*)d_in[6];
  const float* b_out       = (const float*)d_in[7];
  float*       out         = (float*)d_out;

  char* ws = (char*)d_ws;
  size_t off = 0;
  unsigned char* u_all = (unsigned char*)(ws + off); off = align256(off + (size_t)TSTEPS * N_INPUT * BATCH);
  unsigned int*  lp    = (unsigned int*) (ws + off); off = align256(off + (size_t)N_NODES * 128 * 4);
  int*           eff   = (int*)          (ws + off); off = align256(off + (size_t)N_NODES * K_MAX * 4);
  unsigned char* flag  = (unsigned char*)(ws + off); off = align256(off + (size_t)N_NODES);
  unsigned char* sA    = (unsigned char*)(ws + off); off = align256(off + (size_t)(N_NODES + 1) * SROW);
  unsigned char* sB    = (unsigned char*)(ws + off); off = align256(off + (size_t)(N_NODES + 1) * SROW);
  float*         stage = (float*)        (ws + off); off = align256(off + (size_t)BATCH * 16 * 4);
  (void)ws_size; (void)in_sizes; (void)n_in; (void)out_size;

  k_input_wmma<<<256, 256, 0, stream>>>(x, w_in, u_all);
  k_pack_lut  <<<(N_NODES * 128 + 255) / 256, 256, 0, stream>>>(lut, lp);
  k_prep_adj  <<<(N_NODES + 255) / 256, 256, 0, stream>>>(adj_list, adj_mask, eff, flag);
  k_init      <<<((N_NODES + 1) * SROW + 255) / 256, 256, 0, stream>>>(init_states, u_all, sA, sB);
  k_zero_stage<<<4, 256, 0, stream>>>(stage);

  unsigned char* prev = sA;
  unsigned char* cur  = sB;
  for (int t = 0; t < TSTEPS; ++t) {
    k_step<<<(N_NODES * BATCH) / 256, 256, 0, stream>>>(prev, cur, eff, flag, lp, u_all, t);
    unsigned char* tmp = prev; prev = cur; cur = tmp;
  }

  k_readout<<<28, 256, 0, stream>>>(prev, W_out, stage);
  k_final  <<<(BATCH * N_OUT + 255) / 256, 256, 0, stream>>>(stage, b_out, out);
}
